// AdditiveAttention_9199819948223
// MI455X (gfx1250) — matrix operand order repaired and checked by reading, not compiled, not run on hardware
//
#include <hip/hip_runtime.h>

typedef __attribute__((ext_vector_type(16))) _Float16 v16h;
typedef __attribute__((ext_vector_type(8)))  _Float16 v8h;
typedef __attribute__((ext_vector_type(8)))  float    v8f;

#define B_   4
#define Q_   1024
#define K_   1024
#define QS_  512
#define H_   32
#define V_   256
#define KCH  32      // keys per chunk (== WMMA K for f16)
#define PADH 40      // padded LDS row length in halves (80B -> bank stride 20)

// ---------------------------------------------------------------------------
// Kernel 1: projections  qp = queries @ Wq,  kp = keys @ Wk   (tiny: 0.27 GFLOP)
// ---------------------------------------------------------------------------
__global__ __launch_bounds__(256) void proj_kernel(
    const float* __restrict__ queries, const float* __restrict__ keys,
    const float* __restrict__ Wq,      const float* __restrict__ Wk,
    float* __restrict__ qp,            float* __restrict__ kp)
{
  const int lane = threadIdx.x & 31;
  const int row  = blockIdx.x * 8 + (threadIdx.x >> 5);   // 8 waves per block
  const float* in; const float* W; float* outp;
  if (row < B_ * Q_) {
    in = queries + (size_t)row * QS_;          W = Wq; outp = qp + (size_t)row * H_;
  } else {
    const int r2 = row - B_ * Q_;
    in = keys + (size_t)r2 * QS_;              W = Wk; outp = kp + (size_t)r2 * H_;
  }
  float acc = 0.f;
  #pragma unroll 4
  for (int e = 0; e < QS_; ++e)
    acc = fmaf(in[e], W[e * H_ + lane], acc);
  outp[lane] = acc;
}

__device__ __forceinline__ float tanh_fast(float x) {
  // tanh(x) = 1 - 2/(exp(2x)+1); __expf -> v_exp_f32 (TRANS), __fdividef -> v_rcp
  const float e2 = __expf(2.0f * x);
  return 1.0f - __fdividef(2.0f, e2 + 1.0f);
}

// ---------------------------------------------------------------------------
// Kernel 2: fused additive-attention.
// Block = 16 waves (512 thr) owns a (b, 16-query) tile and all 256 V columns.
// Per 32-key chunk: scores -> exp -> LDS f16 P-tile; each wave does one
// v_wmma_f32_16x16x32_f16 of P[16x32] x V[32x16] into its f32 accumulator.
// Unnormalized-softmax accumulation (|score| <= sum|wv| ~ 5, so exp cannot
// overflow and no running max / rescaling is needed); divide by l at the end.
// ---------------------------------------------------------------------------
__global__ __launch_bounds__(512) void attn_kernel(
    const float* __restrict__ values,
    const float* __restrict__ qp, const float* __restrict__ kp,
    const float* __restrict__ wv, float* __restrict__ out)
{
  __shared__ float qp_s[16][H_];          // query projections for this tile
  __shared__ __align__(16) float kp_s[KCH][H_];  // key projections (async-filled)
  __shared__ float l_s[16];               // softmax denominators
  __shared__ __align__(16) _Float16 Ps[16 * PADH];   // exp(scores), 16 x 32 (padded)
  __shared__ __align__(16) _Float16 Vt[V_ * PADH];   // V chunk transposed: [col][k]

  const int tid  = threadIdx.x;
  const int lane = tid & 31;
  const int wave = tid >> 5;              // 0..15 -> output column tile
  const int b    = blockIdx.x >> 6;       // Q_/16 = 64 query tiles per batch
  const int q0   = (blockIdx.x & 63) << 4;

  // one-time tile loads
  {
    const int qi = tid >> 5, h = tid & 31;              // exactly 512 elements
    qp_s[qi][h] = qp[(size_t)(b * Q_ + q0 + qi) * H_ + h];
  }
  if (tid < 16) l_s[tid] = 0.f;

  // wv is uniform across the grid -> read via uniform addresses so the
  // compiler scalarizes it into SGPRs (free VALU operands, no LDS traffic).
  float wvr[H_];
  #pragma unroll
  for (int h = 0; h < H_; ++h) wvr[h] = wv[h];

  __syncthreads();

  // hoist this thread's query row into VGPRs (invariant over all 32 chunks)
  const int qi = tid & 15;                 // this thread's score row
  const int kk = tid >> 4;                 // this thread's score key (0..31)
  float qv[H_];
  #pragma unroll
  for (int h = 0; h < H_; ++h) qv[h] = qp_s[qi][h];

  const int half = lane >> 4;              // lane sub-group within wave
  const int sub  = lane & 15;
  v8f c = {};                              // 16x16 f32 accumulator fragment

  // async-copy setup for kp chunk: contiguous 4KB global -> contiguous LDS
  const unsigned kpLds = (unsigned)(size_t)(&kp_s[0][0]) + (unsigned)tid * 16u;
  const float* kpGBase = kp + (size_t)b * K_ * H_;     // advances by KCH*H_ per chunk

  for (int k0 = 0; k0 < K_; k0 += KCH) {
    __syncthreads();                       // protect LDS from previous-iter readers

    // ---- stage key projections (32x32 f32 = 4KB) via async global->LDS DMA.
    // 256 lanes x 16B. Tracked by ASYNCcnt; consumed after wait + barrier.
    if (tid < 256) {
      const unsigned long long g =
          (unsigned long long)(size_t)(kpGBase + (size_t)(k0 * H_) + tid * 4);
      asm volatile("global_load_async_to_lds_b128 %0, %1, off"
                   :: "v"(kpLds), "v"(g) : "memory");
    }

    // ---- stage V chunk, transposed + f16: Vt[col*PADH + k] (float4 global reads)
    #pragma unroll
    for (int it = 0; it < (KCH * V_ / 4) / 512; ++it) {     // 4 iterations
      const int idx = (tid + it * 512) * 4;                 // element index
      const int kr = idx >> 8, col = idx & 255;             // 4 consecutive cols
      const float4 v4 =
          *(const float4*)&values[(size_t)(b * K_ + k0 + kr) * V_ + col];
      Vt[(col + 0) * PADH + kr] = (_Float16)v4.x;
      Vt[(col + 1) * PADH + kr] = (_Float16)v4.y;
      Vt[(col + 2) * PADH + kr] = (_Float16)v4.z;
      Vt[(col + 3) * PADH + kr] = (_Float16)v4.w;
    }

    asm volatile("s_wait_asynccnt 0x0" ::: "memory");
    __syncthreads();

    // ---- additive score for (qi, kk): s = sum_h wv[h] * tanh(qv[h] + kp[kk][h])
    float s = 0.f;
    const float4* kpv = (const float4*)(&kp_s[kk][0]);      // 8x ds_load_b128
    #pragma unroll
    for (int h4 = 0; h4 < H_ / 4; ++h4) {
      const float4 kq = kpv[h4];
      s = fmaf(wvr[4 * h4 + 0], tanh_fast(qv[4 * h4 + 0] + kq.x), s);
      s = fmaf(wvr[4 * h4 + 1], tanh_fast(qv[4 * h4 + 1] + kq.y), s);
      s = fmaf(wvr[4 * h4 + 2], tanh_fast(qv[4 * h4 + 2] + kq.z), s);
      s = fmaf(wvr[4 * h4 + 3], tanh_fast(qv[4 * h4 + 3] + kq.w), s);
    }
    const float e = __expf(s);             // bounded: |s| <= sum|wv| ~ 5
    Ps[qi * PADH + kk] = (_Float16)e;
    atomicAdd(&l_s[qi], e);                // ds_add_f32
    __syncthreads();

    // --- A fragment: P 16x32 f16, ISA lane layout:
    // lanes 0-15 : row=lane,    K 0..7  (v0-3) and K 16..23 (v4-7)
    // lanes 16-31: row=lane-16, K 8..15 (v0-3) and K 24..31 (v4-7)
    const v8h a0 = *(const v8h*)(Ps + sub * PADH + half * 8);
    const v8h a1 = *(const v8h*)(Ps + sub * PADH + 16 + half * 8);
    const v16h a = __builtin_shufflevector(a0, a1,
        0,1,2,3,4,5,6,7,8,9,10,11,12,13,14,15);

    // --- B fragment: V 32x16 f16 for this wave's column tile:
    // lanes 0-15 : col=base+lane,    K 0..7, 16..23 ; lanes 16-31: col=base+lane-16, K 8..15, 24..31
    const int col = wave * 16 + sub;
    const v8h b0 = *(const v8h*)(Vt + col * PADH + half * 8);
    const v8h b1 = *(const v8h*)(Vt + col * PADH + 16 + half * 8);
    const v16h bm = __builtin_shufflevector(b0, b1,
        0,1,2,3,4,5,6,7,8,9,10,11,12,13,14,15);

    // D = A*B + C  -> v_wmma_f32_16x16x32_f16
    c = __builtin_amdgcn_wmma_f32_16x16x32_f16(
        /*neg_a=*/false, a, /*neg_b=*/false, bm,
        /*c_mod=*/(short)0, c, /*reuse_a=*/false, /*reuse_b=*/false);
  }
  __syncthreads();                         // l_s complete

  // C/D layout: VGPR r -> M = r (lanes 0-15) or r+8 (lanes 16-31), N = lane%16
  // l is a sum of positive exponentials (well-conditioned): native v_rcp_f32
  // + multiply instead of the correctly-rounded division sequence.
  #pragma unroll
  for (int r = 0; r < 8; ++r) {
    const int M   = r + (half << 3);
    const int row = q0 + M;
    const int col = wave * 16 + sub;
    const float inv = __builtin_amdgcn_rcpf(l_s[M]);
    out[(size_t)(b * Q_ + row) * V_ + col] = c[r] * inv;
  }
}

extern "C" void kernel_launch(void* const* d_in, const int* in_sizes, int n_in,
                              void* d_out, int out_size, void* d_ws, size_t ws_size,
                              hipStream_t stream) {
  const float* queries = (const float*)d_in[0];   // [B,Q,512]
  const float* keys    = (const float*)d_in[1];   // [B,K,512]
  const float* values  = (const float*)d_in[2];   // [B,K,256]
  const float* Wq      = (const float*)d_in[3];   // [512,32]
  const float* Wk      = (const float*)d_in[4];   // [512,32]
  const float* wv      = (const float*)d_in[5];   // [32]
  float* out = (float*)d_out;                     // [B,Q,256]

  // workspace: q/k projections (1 MB total)
  float* qpw = (float*)d_ws;                          // [B*Q, 32]
  float* kpw = qpw + (size_t)B_ * Q_ * H_;            // [B*K, 32]

  proj_kernel<<<(B_ * Q_ + B_ * K_) / 8, 256, 0, stream>>>(
      queries, keys, Wq, Wk, qpw, kpw);

  attn_kernel<<<B_ * (Q_ / 16), 512, 0, stream>>>(
      values, qpw, kpw, wv, out);
}
